// make_gradient_47914655154716
// MI455X (gfx1250) — compile-verified, matrix operand order checked by reading, not run on hardware
//
#include <hip/hip_runtime.h>
#include <hip/hip_bf16.h>

// ---------------------------------------------------------------------------
// make_gradient for MI455X (gfx1250, wave32)
//
// Roofline: 33.5 MB in + 201 MB out => ~235 MB compulsory HBM traffic
// => ~10 us at 23.3 TB/s. Memory bound; mask (33.5 MB) stays resident in the
// 192 MB L2, so all neighbor/re-reads are L2 hits. The only serial structure
// is the column-major cumsum; we break it hierarchically and use
// v_wmma_f32_16x16x32_bf16 (scan == lower-triangular-ones matmul; 0/1 data is
// exact in bf16, accumulation in f32) so each wave's critical path is 4 WMMAs
// per 64 rows instead of 64 dependent scalar adds.
//
// v2: all boundary handling is branchless (clamped addresses + 0/1 value
// weights) so the hot emission loop has no exec-mask divergence / saveexec
// branches; K1/K4 vectorized to b128 accesses.
// ---------------------------------------------------------------------------

typedef __attribute__((ext_vector_type(16))) __bf16 v16bf;
typedef __attribute__((ext_vector_type(8)))  float  v8f;

#define B_      32
#define H_      512
#define W_      512
#define CHUNKS  8
#define CH_ROWS 64   // H_/CHUNKS

// K1: partial column sums PS[b][chunk][c] = sum of mask over the chunk's rows.
// 128 threads * float4 = 512 columns per block, coalesced b128 loads.
__global__ void k1_colsum(const float* __restrict__ mask, float* __restrict__ ps) {
    int b  = blockIdx.x >> 3;
    int ch = blockIdx.x & 7;
    int c4 = threadIdx.x;                       // 128 threads, 4 columns each
    const float4* p = (const float4*)(mask + (size_t)(b * H_ + ch * CH_ROWS) * W_) + c4;
    float4 s = make_float4(0.f, 0.f, 0.f, 0.f);
    #pragma unroll 8
    for (int i = 0; i < CH_ROWS; ++i) {
        float4 v = p[(size_t)i * (W_ / 4)];
        s.x += v.x; s.y += v.y; s.z += v.z; s.w += v.w;
    }
    ((float4*)(ps + (b * CHUNKS + ch) * W_))[c4] = s;
}

// K2: per batch, turn PS into carry-in for each (chunk, column):
//   carry[b][ch][c] = (ones in all columns < c) + (ones in rows above chunk, col c)
// Exclusive column scan via Hillis-Steele in LDS; rewritten in place.
__global__ void k2_scan(float* __restrict__ ps) {
    __shared__ float sm[W_];
    int b = blockIdx.x;
    int c = threadIdx.x;
    float p[CHUNKS];
    float tot = 0.f;
    #pragma unroll
    for (int j = 0; j < CHUNKS; ++j) { p[j] = ps[(b * CHUNKS + j) * W_ + c]; tot += p[j]; }
    sm[c] = tot;
    __syncthreads();
    #pragma unroll
    for (int off = 1; off < W_; off <<= 1) {
        float v = (c >= off) ? sm[c - off] : 0.f;
        __syncthreads();
        sm[c] += v;
        __syncthreads();
    }
    float run = sm[c] - tot;                    // exclusive prefix over columns
    #pragma unroll
    for (int j = 0; j < CHUNKS; ++j) { ps[(b * CHUNKS + j) * W_ + c] = run; run += p[j]; }
}

// K3: fused scan (WMMA) + Omega + index_matrix.
// One wave per (batch, 64-row chunk, 16-column tile): 32*8*32 = 8192 waves.
// Per 32-row group: S = L * X + carry using two v_wmma_f32_16x16x32_bf16.
//
// bf16 operand layout (ISA 7.12.2, 16-bit A 16x32 mirrored for B):
//   element e of v16bf, lane-half hi: v=e>>1, j=e&1,
//   K = 2*(v&3) + j + (v>=4 ? 16 : 0) + (hi ? 8 : 0)
//   A: lanes hold M (lane&15); B: lanes hold N (lane&15).
// C/D (f32 16x16): reg j, lane l -> M = j + 8*hi, N = lane&15.
__global__ void k3_scan_wmma(const float* __restrict__ mask,
                             const float* __restrict__ carryBuf,
                             float* __restrict__ omega,
                             float* __restrict__ indexm) {
    const int  lane = threadIdx.x & 31;
    const int  wave = threadIdx.x >> 5;
    int bt = blockIdx.x;
    const int tg = bt & 3;  bt >>= 2;           // 4 tile-groups of 8 waves
    const int ch = bt & 7;  bt >>= 3;           // 8 row chunks
    const int b  = bt;                          // 32 batches
    const int c0 = (tg * 8 + wave) * 16;        // 16-column tile base
    const int n  = lane & 15;
    const int hi = (lane >= 16) ? 1 : 0;
    const int m  = n;                           // A-matrix row for this lane
    const int r0 = ch * CH_ROWS;
    const int c  = c0 + n;

    // Per-lane, loop-invariant column-boundary handling (branchless).
    const int   offl = (c > 0)      ? -1 : 0;
    const float wl   = (c > 0)      ? 1.f : 0.f;
    const int   offr = (c + 1 < W_) ?  1 : 0;
    const float wr   = (c + 1 < W_) ? 1.f : 0.f;

    // Lower-triangular scan matrices: A1 -> rows 0..15 of S, A2 -> rows 16..31.
    v16bf a1, a2;
    #pragma unroll
    for (int e = 0; e < 16; ++e) {
        int v = e >> 1, j = e & 1;
        int k = ((v & 3) << 1) + j + ((v >= 4) ? 16 : 0) + (hi ? 8 : 0);
        a1[e] = (__bf16)((k <= m)      ? 1.0f : 0.0f);
        a2[e] = (__bf16)((k <= m + 16) ? 1.0f : 0.0f);
    }

    float carry = carryBuf[(b * CHUNKS + ch) * W_ + c0 + n];
    const size_t imgBase = (size_t)b * H_ * W_;

    #pragma unroll
    for (int g = 0; g < CH_ROWS / 32; ++g) {
        const int rg = r0 + g * 32;

        // Load 32x16 mask tile into B layout, converted to bf16 (0/1 exact).
        v16bf bm;
        #pragma unroll
        for (int e = 0; e < 16; ++e) {
            int v = e >> 1, j = e & 1;
            int k = ((v & 3) << 1) + j + ((v >= 4) ? 16 : 0) + (hi ? 8 : 0);
            bm[e] = (__bf16)mask[imgBase + (size_t)(rg + k) * W_ + c0 + n];
        }

        v8f cacc;
        #pragma unroll
        for (int j = 0; j < 8; ++j) cacc[j] = carry;   // carry broadcast per column

        v8f d1 = __builtin_amdgcn_wmma_f32_16x16x32_bf16(false, a1, false, bm,
                                                         (short)0, cacc, false, false);
        v8f d2 = __builtin_amdgcn_wmma_f32_16x16x32_bf16(false, a2, false, bm,
                                                         (short)0, cacc, false, false);

        // Next group's carry = S[row 31] = d2 reg7, lanes 16..31 (M=15).
        carry = __shfl(d2[7], 16 + n, 32);

        // Emit index_matrix and Omega for these 32 rows (D layout addressing).
        // All neighbor loads are unconditional (clamped offsets); boundary
        // zeros come from 0/1 value weights -> no exec divergence.
        #pragma unroll
        for (int j = 0; j < 8; ++j) {
            #pragma unroll
            for (int half = 0; half < 2; ++half) {
                int   r   = rg + j + hi * 8 + half * 16;
                float s   = half ? d2[j] : d1[j];
                size_t pix = imgBase + (size_t)r * W_ + c;

                const int   offd = (r + 1 < H_) ? W_  : 0;
                const float wd   = (r + 1 < H_) ? 1.f : 0.f;
                const int   offu = (r > 0)      ? -W_ : 0;
                const float wu   = (r > 0)      ? 1.f : 0.f;

                float mc = mask[pix];                    // L0/L2 hit
                float md = mask[pix + offd] * wd;
                float mu = mask[pix + offu] * wu;
                float mr = mask[pix + offr] * wr;
                float ml = mask[pix + offl] * wl;

                ((float4*)omega)[pix] = make_float4(mc * md, mc * mu, mc * mr, mc * ml);
                indexm[pix] = mc * s;
            }
        }
    }
}

// K4: right_idx = mask * index_matrix[r][c+1]  (mask^2 == mask for 0/1 data).
// Separate pass: the c+1 value at a tile boundary is produced by a different
// wave in K3; stream ordering makes this race-free, and indexm is L2-hot.
// 4 pixels/thread with b128 loads/stores; shifted tail element handled with an
// unconditional clamped load + value select (branchless).
__global__ void k4_rightidx(const float* __restrict__ mask,
                            const float* __restrict__ indexm,
                            float* __restrict__ ridx) {
    size_t tid = (size_t)blockIdx.x * blockDim.x + threadIdx.x;
    size_t t4  = tid * 4;
    int    cq  = (int)(t4 & (W_ - 1));            // column of first element (mult of 4)
    float4 m4  = ((const float4*)mask)[tid];
    float4 i4  = ((const float4*)indexm)[tid];
    const int   nadd = (cq + 4 < W_) ? 4   : 0;   // clamped -> load always valid
    const float nw   = (cq + 4 < W_) ? 1.f : 0.f;
    float inx = indexm[t4 + nadd] * nw;           // index_matrix[r][cq+4] or 0
    float4 o;
    o.x = m4.x * i4.y;
    o.y = m4.y * i4.z;
    o.z = m4.z * i4.w;
    o.w = m4.w * inx;
    ((float4*)ridx)[tid] = o;
}

extern "C" void kernel_launch(void* const* d_in, const int* in_sizes, int n_in,
                              void* d_out, int out_size, void* d_ws, size_t ws_size,
                              hipStream_t stream) {
    (void)in_sizes; (void)n_in; (void)out_size; (void)ws_size;
    const float* mask = (const float*)d_in[0];
    float* out    = (float*)d_out;
    float* omega  = out;                                   // [B,H,W,4]
    float* indexm = out + (size_t)B_ * H_ * W_ * 4;        // [B,H,W]
    float* ridx   = indexm + (size_t)B_ * H_ * W_;         // [B,H,W]
    float* ps     = (float*)d_ws;                          // B*CHUNKS*W f32 = 512 KB

    k1_colsum   <<<B_ * CHUNKS,             128, 0, stream>>>(mask, ps);
    k2_scan     <<<B_,                      W_,  0, stream>>>(ps);
    k3_scan_wmma<<<B_ * CHUNKS * 4,         256, 0, stream>>>(mask, ps, omega, indexm);
    k4_rightidx <<<(B_ * H_ * W_) / (256*4), 256, 0, stream>>>(mask, indexm, ridx);
}
